// MelGAT_29583734734922
// MI455X (gfx1250) — matrix operand order repaired and checked by reading, not compiled, not run on hardware
//
#include <hip/hip_runtime.h>
#include <hip/hip_bf16.h>
#include <math.h>

typedef __attribute__((ext_vector_type(16))) __bf16          v16bf;
typedef __attribute__((ext_vector_type(8)))  float           v8f;
typedef __attribute__((ext_vector_type(8)))  unsigned short  ush8;

#define DIN   1024
#define HID   128
#define HEADS 2
#define NC    4

// ---------- helpers ----------
__device__ __forceinline__ unsigned short f2bf(float f) {
    unsigned u = __float_as_uint(f);
    u += 0x7fffu + ((u >> 16) & 1u);          // round-to-nearest-even
    return (unsigned short)(u >> 16);
}
__device__ __forceinline__ float bf2f(unsigned short h) {
    return __uint_as_float(((unsigned)h) << 16);
}
__device__ __forceinline__ void atomicMaxF(float* a, float v) {
    // sign-punned monotone mapping: works when initialized to -inf
    if (v >= 0.0f) atomicMax((int*)a, __float_as_int(v));
    else           atomicMin((unsigned int*)a, (unsigned int)__float_as_int(v));
}

union Frag { v16bf v; ush8 u[2]; };

// =====================================================================
// Kernel 1: h = relu(x @ W_pre + b_pre), output bf16 [N,128]
// bf16 WMMA, fp32->bf16 convert on the fly, 128x128 tile per block.
// =====================================================================
__global__ __launch_bounds__(256) void k_gemm_pre(
    const float* __restrict__ x, const float* __restrict__ Wp,
    const float* __restrict__ bp, unsigned short* __restrict__ hbf, int nrows)
{
    __shared__ unsigned short lA[128 * 40];   // 128 rows x 32 K (stride 40: aligned + conflict-free)
    __shared__ unsigned short lB[32 * 136];   // 32 K x 128 cols (stride 136)

    const int tid = threadIdx.x, lane = tid & 31, wv = tid >> 5;
    const long Rb0 = (long)blockIdx.x * 128;

    v8f acc[8];
#pragma unroll
    for (int n = 0; n < 8; ++n)
#pragma unroll
        for (int e = 0; e < 8; ++e) acc[n][e] = 0.0f;

    const int ar = tid >> 1, ah = tid & 1;          // A loader: row, K-half
    const long grow = Rb0 + ar;
    const int bk = tid >> 3, bs = tid & 7;          // B loader: K row, 16-col segment
    const int cl = lane & 15, sel = lane >> 4;

    for (int k0 = 0; k0 < DIN; k0 += 32) {
        // ---- A tile (fp32 -> bf16) ----
        {
            float va[16];
            if (grow < nrows) {
                const float4* p = (const float4*)(x + grow * DIN + k0 + ah * 16);
#pragma unroll
                for (int q = 0; q < 4; ++q) {
                    float4 t = p[q];
                    va[q*4+0]=t.x; va[q*4+1]=t.y; va[q*4+2]=t.z; va[q*4+3]=t.w;
                }
                if (k0 + 32 < DIN)
                    __builtin_prefetch(x + grow * DIN + k0 + 32 + ah * 16, 0, 1);
            } else {
#pragma unroll
                for (int q = 0; q < 16; ++q) va[q] = 0.0f;
            }
            ush8 p0, p1;
#pragma unroll
            for (int q = 0; q < 8; ++q) { p0[q] = f2bf(va[q]); p1[q] = f2bf(va[8+q]); }
            *(ush8*)&lA[ar * 40 + ah * 16]     = p0;
            *(ush8*)&lA[ar * 40 + ah * 16 + 8] = p1;
        }
        // ---- B tile (fp32 -> bf16), K-row / N-contiguous (native WMMA-B layout) ----
        {
            const float4* p = (const float4*)(Wp + (size_t)(k0 + bk) * HID + bs * 16);
            float vb[16];
#pragma unroll
            for (int q = 0; q < 4; ++q) {
                float4 t = p[q];
                vb[q*4+0]=t.x; vb[q*4+1]=t.y; vb[q*4+2]=t.z; vb[q*4+3]=t.w;
            }
            ush8 q0, q1;
#pragma unroll
            for (int q = 0; q < 8; ++q) { q0[q] = f2bf(vb[q]); q1[q] = f2bf(vb[8+q]); }
            *(ush8*)&lB[bk * 136 + bs * 16]     = q0;
            *(ush8*)&lB[bk * 136 + bs * 16 + 8] = q1;
        }
        __syncthreads();

        // A fragment: lane<16 -> K{0..7,16..23}; lane>=16 -> K{8..15,24..31}
        Frag fa;
        const unsigned short* arow = &lA[(wv * 16 + cl) * 40 + sel * 8];
        fa.u[0] = *(const ush8*)(arow);
        fa.u[1] = *(const ush8*)(arow + 16);
#pragma unroll
        for (int n = 0; n < 8; ++n) {
            Frag fb;  // B fragment: lane = N column, K in same order as A
            const unsigned short* bcol = &lB[n * 16 + cl];
#pragma unroll
            for (int j = 0; j < 8; ++j) {
                fb.u[0][j] = bcol[(sel * 8 + j) * 136];
                fb.u[1][j] = bcol[(16 + sel * 8 + j) * 136];
            }
            acc[n] = __builtin_amdgcn_wmma_f32_16x16x32_bf16(
                false, fa.v, false, fb.v, (short)0, acc[n], false, false);
        }
        __syncthreads();
    }

    // epilogue: + bias, relu, store bf16
    const long rbase = Rb0 + wv * 16 + sel * 8;
#pragma unroll
    for (int n = 0; n < 8; ++n) {
        float bias = bp[n * 16 + cl];
#pragma unroll
        for (int r = 0; r < 8; ++r) {
            long row = rbase + r;
            if (row < nrows) {
                float v = acc[n][r] + bias;
                v = v > 0.0f ? v : 0.0f;
                hbf[row * HID + n * 16 + cl] = f2bf(v);
            }
        }
    }
}

// =====================================================================
// Kernel 2: h2 = h @ W1 (no bias pre-attention), output bf16 [N,256];
// fused alpha_src/alpha_dst reductions in WMMA epilogue.
// blockIdx.y = head (column half).
// =====================================================================
__global__ __launch_bounds__(256) void k_gemm_gat1(
    const unsigned short* __restrict__ hbf, const float* __restrict__ W1,
    const float* __restrict__ asrc, const float* __restrict__ adst,
    unsigned short* __restrict__ h2bf, float* __restrict__ als,
    float* __restrict__ ald, int nrows)
{
    __shared__ unsigned short lA[128 * 40];
    __shared__ unsigned short lB[32 * 136];

    const int tid = threadIdx.x, lane = tid & 31, wv = tid >> 5;
    const long Rb0 = (long)blockIdx.x * 128;
    const int head = blockIdx.y, c0 = head * HID;

    v8f acc[8];
#pragma unroll
    for (int n = 0; n < 8; ++n)
#pragma unroll
        for (int e = 0; e < 8; ++e) acc[n][e] = 0.0f;

    const int ar = tid >> 1, ah = tid & 1;
    const long grow = Rb0 + ar;
    const int bk = tid >> 3, bs = tid & 7;
    const int cl = lane & 15, sel = lane >> 4;

    for (int k0 = 0; k0 < HID; k0 += 32) {
        // A tile: already bf16 in global, straight copy
        {
            ush8 p0 = {}, p1 = {};
            if (grow < nrows) {
                const ush8* p = (const ush8*)(hbf + grow * HID + k0 + ah * 16);
                p0 = p[0]; p1 = p[1];
            }
            *(ush8*)&lA[ar * 40 + ah * 16]     = p0;
            *(ush8*)&lA[ar * 40 + ah * 16 + 8] = p1;
        }
        // B tile: W1[k0+bk][c0 + bs*16 ..], fp32 -> bf16
        {
            const float4* p = (const float4*)(W1 + (size_t)(k0 + bk) * (HEADS * HID) + c0 + bs * 16);
            float vb[16];
#pragma unroll
            for (int q = 0; q < 4; ++q) {
                float4 t = p[q];
                vb[q*4+0]=t.x; vb[q*4+1]=t.y; vb[q*4+2]=t.z; vb[q*4+3]=t.w;
            }
            ush8 q0, q1;
#pragma unroll
            for (int q = 0; q < 8; ++q) { q0[q] = f2bf(vb[q]); q1[q] = f2bf(vb[8+q]); }
            *(ush8*)&lB[bk * 136 + bs * 16]     = q0;
            *(ush8*)&lB[bk * 136 + bs * 16 + 8] = q1;
        }
        __syncthreads();

        Frag fa;
        const unsigned short* arow = &lA[(wv * 16 + cl) * 40 + sel * 8];
        fa.u[0] = *(const ush8*)(arow);
        fa.u[1] = *(const ush8*)(arow + 16);
#pragma unroll
        for (int n = 0; n < 8; ++n) {
            Frag fb;
            const unsigned short* bcol = &lB[n * 16 + cl];
#pragma unroll
            for (int j = 0; j < 8; ++j) {
                fb.u[0][j] = bcol[(sel * 8 + j) * 136];
                fb.u[1][j] = bcol[(16 + sel * 8 + j) * 136];
            }
            acc[n] = __builtin_amdgcn_wmma_f32_16x16x32_bf16(
                false, fa.v, false, fb.v, (short)0, acc[n], false, false);
        }
        __syncthreads();
    }

    // epilogue: store h2 (bf16) + fused per-row attention-score reductions
    const long rbase = Rb0 + wv * 16 + sel * 8;
    float ps[8], pd[8];
#pragma unroll
    for (int r = 0; r < 8; ++r) { ps[r] = 0.0f; pd[r] = 0.0f; }

#pragma unroll
    for (int n = 0; n < 8; ++n) {
        float wa = asrc[head * HID + n * 16 + cl];
        float wd = adst[head * HID + n * 16 + cl];
#pragma unroll
        for (int r = 0; r < 8; ++r) {
            float v = acc[n][r];
            ps[r] += v * wa;
            pd[r] += v * wd;
            long row = rbase + r;
            if (row < nrows)
                h2bf[row * (HEADS * HID) + c0 + n * 16 + cl] = f2bf(v);
        }
    }
    // reduce across the 16 column lanes (stays within each 16-lane half)
#pragma unroll
    for (int m = 1; m <= 8; m <<= 1)
#pragma unroll
        for (int r = 0; r < 8; ++r) {
            ps[r] += __shfl_xor(ps[r], m);
            pd[r] += __shfl_xor(pd[r], m);
        }
    if (cl == 0) {
#pragma unroll
        for (int r = 0; r < 8; ++r) {
            long row = rbase + r;
            if (row < nrows) {
                als[row * HEADS + head] = ps[r];
                ald[row * HEADS + head] = pd[r];
            }
        }
    }
}

// =====================================================================
// Edge softmax passes (generic over H heads)
// =====================================================================
__global__ __launch_bounds__(256) void k_edge_max(
    const long long* __restrict__ ei, const float* __restrict__ als,
    const float* __restrict__ ald, float* __restrict__ mx,
    int E, int TE, int H)
{
    int t = blockIdx.x * 256 + threadIdx.x;
    if (t >= TE * H) return;
    int e = t / H, h = t % H;
    int s, d;
    if (e < E) { s = (int)ei[e]; d = (int)ei[E + e]; } else { s = e - E; d = s; }
    float v = als[s * H + h] + ald[d * H + h];
    v = v > 0.0f ? v : 0.2f * v;
    atomicMaxF(&mx[d * H + h], v);
}

__global__ __launch_bounds__(256) void k_edge_sum(
    const long long* __restrict__ ei, const float* __restrict__ als,
    const float* __restrict__ ald, const float* __restrict__ mx,
    float* __restrict__ sm, int E, int TE, int H)
{
    int t = blockIdx.x * 256 + threadIdx.x;
    if (t >= TE * H) return;
    int e = t / H, h = t % H;
    int s, d;
    if (e < E) { s = (int)ei[e]; d = (int)ei[E + e]; } else { s = e - E; d = s; }
    float v = als[s * H + h] + ald[d * H + h];
    v = v > 0.0f ? v : 0.2f * v;
    atomicAdd(&sm[d * H + h], expf(v - mx[d * H + h]));
}

// GAT-1 message scatter: one block per edge, 256 channels (2 heads x 128)
__global__ __launch_bounds__(256) void k_scatter1(
    const long long* __restrict__ ei, const unsigned short* __restrict__ h2bf,
    const float* __restrict__ als, const float* __restrict__ ald,
    const float* __restrict__ mx, const float* __restrict__ sm,
    float* __restrict__ out1, int E)
{
    int e = blockIdx.x;
    int c = threadIdx.x;
    int head = c >> 7;
    int lane = c & 31;
    int s, d;
    if (e < E) { s = (int)ei[e]; d = (int)ei[E + e]; } else { s = e - E; d = s; }
    float coef = 0.0f;
    if (lane == 0) {
        float v = als[s * HEADS + head] + ald[d * HEADS + head];
        v = v > 0.0f ? v : 0.2f * v;
        coef = expf(v - mx[d * HEADS + head]) / (sm[d * HEADS + head] + 1e-16f);
    }
    coef = __shfl(coef, 0);
    float msg = coef * bf2f(h2bf[(long)s * (HEADS * HID) + c]);
    atomicAdd(&out1[(long)d * (HEADS * HID) + c], msg);
}

// =====================================================================
// GAT-2 linear: hl2 = elu(out1 + b1) @ W2 (256->4) + alpha2 scores.
// One wave per node, tree reduce via shfl_xor.
// =====================================================================
__global__ __launch_bounds__(256) void k_gat2_lin(
    const float* __restrict__ out1, const float* __restrict__ b1,
    const float* __restrict__ W2, const float* __restrict__ as2,
    const float* __restrict__ ad2, float* __restrict__ hl2,
    float* __restrict__ als, float* __restrict__ ald, int n)
{
    int wv = threadIdx.x >> 5, lane = threadIdx.x & 31;
    long node = (long)blockIdx.x * 8 + wv;
    if (node >= n) return;
    float a[4] = {0.f, 0.f, 0.f, 0.f};
#pragma unroll
    for (int j = 0; j < 8; ++j) {
        int c = j * 32 + lane;
        float h = out1[node * (HEADS * HID) + c] + b1[c];
        h = h > 0.0f ? h : (expf(h) - 1.0f);
        const float* w = W2 + c * NC;
        a[0] += h * w[0]; a[1] += h * w[1]; a[2] += h * w[2]; a[3] += h * w[3];
    }
#pragma unroll
    for (int m = 16; m > 0; m >>= 1)
#pragma unroll
        for (int q = 0; q < 4; ++q) a[q] += __shfl_xor(a[q], m);
    if (lane == 0) {
        float sa = 0.f, da = 0.f;
#pragma unroll
        for (int q = 0; q < 4; ++q) {
            hl2[node * NC + q] = a[q];
            sa += a[q] * as2[q];
            da += a[q] * ad2[q];
        }
        als[node] = sa;
        ald[node] = da;
    }
}

// GAT-2 message scatter: thread per edge, 4 channels
__global__ __launch_bounds__(256) void k_scatter2(
    const long long* __restrict__ ei, const float* __restrict__ hl2,
    const float* __restrict__ als, const float* __restrict__ ald,
    const float* __restrict__ mx, const float* __restrict__ sm,
    float* __restrict__ out, int E, int TE)
{
    int e = blockIdx.x * 256 + threadIdx.x;
    if (e >= TE) return;
    int s, d;
    if (e < E) { s = (int)ei[e]; d = (int)ei[E + e]; } else { s = e - E; d = s; }
    float v = als[s] + ald[d];
    v = v > 0.0f ? v : 0.2f * v;
    float coef = expf(v - mx[d]) / (sm[d] + 1e-16f);
#pragma unroll
    for (int c = 0; c < 4; ++c)
        atomicAdd(&out[(long)d * NC + c], coef * hl2[(long)s * NC + c]);
}

// ---------- init kernels ----------
__global__ __launch_bounds__(256) void k_fill(float* __restrict__ p, long n, float v)
{
    long i = (long)blockIdx.x * 256 + threadIdx.x;
    if (i < n) p[i] = v;
}
__global__ __launch_bounds__(256) void k_fill_b2(float* __restrict__ out,
                                                 const float* __restrict__ b2, int n)
{
    int i = blockIdx.x * 256 + threadIdx.x;
    if (i < n * NC) out[i] = b2[i & 3];
}

// =====================================================================
extern "C" void kernel_launch(void* const* d_in, const int* in_sizes, int n_in,
                              void* d_out, int out_size, void* d_ws, size_t ws_size,
                              hipStream_t stream)
{
    const float*     x   = (const float*)d_in[0];
    const long long* ei  = (const long long*)d_in[1];   // int64 per reference
    const float*     Wp  = (const float*)d_in[2];
    const float*     bp  = (const float*)d_in[3];
    const float*     W1  = (const float*)d_in[4];
    const float*     as1 = (const float*)d_in[5];
    const float*     ad1 = (const float*)d_in[6];
    const float*     b1  = (const float*)d_in[7];
    const float*     W2  = (const float*)d_in[8];
    const float*     as2 = (const float*)d_in[9];
    const float*     ad2 = (const float*)d_in[10];
    const float*     b2  = (const float*)d_in[11];

    const int N  = in_sizes[0] / DIN;
    const int E  = in_sizes[1] / 2;
    const int TE = E + N;

    // workspace carve (256B aligned)
    unsigned char* w = (unsigned char*)d_ws;
    auto carve = [&](size_t bytes) -> void* {
        void* p = (void*)w;
        w += (bytes + 255) & ~(size_t)255;
        return p;
    };
    unsigned short* hbf   = (unsigned short*)carve((size_t)N * HID * 2);
    unsigned short* h2bf  = (unsigned short*)carve((size_t)N * HEADS * HID * 2);
    float* als1v = (float*)carve((size_t)N * HEADS * 4);
    float* ald1v = (float*)carve((size_t)N * HEADS * 4);
    float* m1    = (float*)carve((size_t)N * HEADS * 4);
    float* s1    = (float*)carve((size_t)N * HEADS * 4);
    float* out1  = (float*)carve((size_t)N * HEADS * HID * 4);
    float* hl2   = (float*)carve((size_t)N * NC * 4);
    float* als2v = (float*)carve((size_t)N * 4);
    float* ald2v = (float*)carve((size_t)N * 4);
    float* m2    = (float*)carve((size_t)N * 4);
    float* s2    = (float*)carve((size_t)N * 4);

    dim3 blk(256);
    const float NEGINF = -INFINITY;

    // Stage 1: h = relu(x @ W_pre + b_pre)   [WMMA bf16]
    k_gemm_pre<<<dim3((N + 127) / 128), blk, 0, stream>>>(x, Wp, bp, hbf, N);

    // Stage 2: h2 = h @ W1, fused alpha reductions   [WMMA bf16]
    k_gemm_gat1<<<dim3((N + 127) / 128, HEADS), blk, 0, stream>>>(
        hbf, W1, as1, ad1, h2bf, als1v, ald1v, N);

    // GAT-1 segment softmax + scatter
    long n2 = (long)N * HEADS;
    k_fill<<<(unsigned)((n2 + 255) / 256), blk, 0, stream>>>(m1, n2, NEGINF);
    k_fill<<<(unsigned)((n2 + 255) / 256), blk, 0, stream>>>(s1, n2, 0.0f);
    long no = (long)N * HEADS * HID;
    k_fill<<<(unsigned)((no + 255) / 256), blk, 0, stream>>>(out1, no, 0.0f);

    int tw = TE * HEADS;
    k_edge_max<<<(tw + 255) / 256, blk, 0, stream>>>(ei, als1v, ald1v, m1, E, TE, HEADS);
    k_edge_sum<<<(tw + 255) / 256, blk, 0, stream>>>(ei, als1v, ald1v, m1, s1, E, TE, HEADS);
    k_scatter1<<<TE, blk, 0, stream>>>(ei, h2bf, als1v, ald1v, m1, s1, out1, E);

    // GAT-2 linear (elu(out1 + b1) @ W2) + alpha2
    k_gat2_lin<<<(N + 7) / 8, blk, 0, stream>>>(out1, b1, W2, as2, ad2,
                                                hl2, als2v, ald2v, N);

    // GAT-2 segment softmax + scatter into d_out (pre-filled with b2)
    k_fill<<<(N + 255) / 256, blk, 0, stream>>>(m2, (long)N, NEGINF);
    k_fill<<<(N + 255) / 256, blk, 0, stream>>>(s2, (long)N, 0.0f);
    k_fill_b2<<<(N * NC + 255) / 256, blk, 0, stream>>>((float*)d_out, b2, N);

    k_edge_max<<<(TE + 255) / 256, blk, 0, stream>>>(ei, als2v, ald2v, m2, E, TE, 1);
    k_edge_sum<<<(TE + 255) / 256, blk, 0, stream>>>(ei, als2v, ald2v, m2, s2, E, TE, 1);
    k_scatter2<<<(TE + 255) / 256, blk, 0, stream>>>(ei, hl2, als2v, ald2v, m2, s2,
                                                    (float*)d_out, E, TE);
}
